// Transducerv2_26096221291274
// MI455X (gfx1250) — matrix operand order repaired and checked by reading, not compiled, not run on hardware
//
#include <hip/hip_runtime.h>
#include <hip/hip_bf16.h>

// ---------------------------------------------------------------------------
// RNN-Transducer for MI455X (gfx1250):
//   * all GEMMs on the matrix pipe (v_wmma_f32_16x16x32_f16, f32 accumulate)
//   * branch-free double-buffered LDS staging via global_load_async_to_lds_b128
//     (ASYNCcnt) or the Tensor Data Mover (tensor_load_to_lds, TENSORcnt)
//   * 64x128 block tile, 16x64 wave tile (4 WMMA per K-step per wave)
// ---------------------------------------------------------------------------

typedef __attribute__((ext_vector_type(16))) _Float16 v16h;
typedef __attribute__((ext_vector_type(8)))  _Float16 v8h;
typedef __attribute__((ext_vector_type(8)))  float    v8f;
typedef __attribute__((ext_vector_type(4)))  unsigned int u32x4;
typedef __attribute__((ext_vector_type(8)))  int      i32x8;
typedef __attribute__((ext_vector_type(4)))  int      i32x4;

#define WMMA_BM 64
#define WMMA_BN 128
#define WMMA_BK 32
#define LDT 40   // LDS row pitch in halves (32 data + 8 pad)

__device__ __forceinline__ float sigmoidf_(float x) { return 1.f / (1.f + __expf(-x)); }
__device__ __forceinline__ int imin_(int a, int b) { return a < b ? a : b; }

// ---- CDNA5 async copy: global -> LDS, 16B per lane, tracked by ASYNCcnt ----
__device__ __forceinline__ void async_ld_b128(unsigned int lds_off, const void* gp) {
  asm volatile("global_load_async_to_lds_b128 %0, %1, off"
               :: "v"(lds_off), "v"((unsigned long long)(uintptr_t)gp)
               : "memory");
}
__device__ __forceinline__ void wait_async0() {
  asm volatile("s_wait_asynccnt 0x0" ::: "memory");
}

// ---- small utility kernels -------------------------------------------------

__global__ void __launch_bounds__(256)
k_cvt_f32_f16(const float* __restrict__ in, _Float16* __restrict__ out, int n) {
  int i = blockIdx.x * blockDim.x + threadIdx.x;
  if (i < n) out[i] = (_Float16)in[i];
}

// convert rows of width Kin into rows of width Kout (zero-padded tail)
__global__ void __launch_bounds__(256)
k_cvt_pad_f16(const float* __restrict__ in, _Float16* __restrict__ out,
              int rows, int Kin, int Kout) {
  int i = blockIdx.x * blockDim.x + threadIdx.x;
  if (i >= rows * Kout) return;
  int r = i / Kout, k = i % Kout;
  out[i] = (k < Kin) ? (_Float16)in[(size_t)r * Kin + k] : (_Float16)0.f;
}

__global__ void __launch_bounds__(256)
k_add2_f32(const float* __restrict__ a, const float* __restrict__ b,
           float* __restrict__ o, int n) {
  int i = blockIdx.x * blockDim.x + threadIdx.x;
  if (i < n) o[i] = a[i] + b[i];
}

__global__ void __launch_bounds__(256)
k_zero_f16(_Float16* p, int n) {
  int i = blockIdx.x * blockDim.x + threadIdx.x;
  if (i < n) p[i] = (_Float16)0.f;
}

__global__ void __launch_bounds__(256)
k_zero_f32(float* p, int n) {
  int i = blockIdx.x * blockDim.x + threadIdx.x;
  if (i < n) p[i] = 0.f;
}

// ---- fragment helpers ------------------------------------------------------

__device__ __forceinline__ v16h load_a_frag(const _Float16* ar, int lh) {
  // 16-bit A 16x32: lanes 0-15 hold K 0..7 / 16..23, lanes 16-31 K 8..15 / 24..31
  const int kb = lh * 8;
  v8h lo = *(const v8h*)&ar[kb];
  v8h hi = *(const v8h*)&ar[16 + kb];
  v16h f;
#pragma unroll
  for (int i = 0; i < 8; ++i) { f[i] = lo[i]; f[8 + i] = hi[i]; }
  return f;
}

__device__ __forceinline__ v16h load_b_frag(const _Float16* wr, int lh) {
  // 16-bit B 32x16: lanes 0-15 hold K 0..7 / 16..23, lanes 16-31 K 8..15 / 24..31
  const int kb = lh * 8;
  v8h lo = *(const v8h*)&wr[kb];
  v8h hi = *(const v8h*)&wr[16 + kb];
  v16h f;
#pragma unroll
  for (int i = 0; i < 8; ++i) { f[i] = lo[i]; f[8 + i] = hi[i]; }
  return f;
}

// ---- TDM descriptor issue (2-D tile; DMA zero-fills past tensor_dim) -------

#if __has_builtin(__builtin_amdgcn_tensor_load_to_lds)
#define HAVE_TDM 1
__device__ __forceinline__ void tdm_load_tile(unsigned int lds_addr, const _Float16* gtile,
                                              int rem_k, int rem_rows, int row_stride,
                                              int tile_rows) {
  const unsigned long long ga = (unsigned long long)(uintptr_t)gtile;
  u32x4 g0;
  g0[0] = 1u;                                   // count=1 (valid user descriptor)
  g0[1] = lds_addr;                             // lds_addr (bytes)
  g0[2] = (unsigned int)ga;                     // global_addr[31:0]
  g0[3] = (unsigned int)((ga >> 32) & 0x01FFFFFFu) | (2u << 30);  // addr[56:32]|type=2
  i32x8 g1;
  g1[0] = (1 << 16)                             // data_size = 2 bytes
        | (1 << 20) | (3 << 22) | (3 << 25);    // pad: every 16 dwords add 4 dwords
  g1[1] = (rem_k & 0xFFFF) << 16;               // tensor_dim0 lo
  g1[2] = ((unsigned)rem_k >> 16) | ((rem_rows & 0xFFFF) << 16);  // dim0 hi | dim1 lo
  g1[3] = ((unsigned)rem_rows >> 16) | (WMMA_BK << 16);           // dim1 hi | tile_dim0
  g1[4] = tile_rows;                            // tile_dim1 (tile_dim2 = 0)
  g1[5] = row_stride;                           // tensor_dim0_stride lo
  g1[6] = 0;
  g1[7] = 0;
  i32x4 g2 = {};
  i32x4 g3 = {};
#if defined(__clang_major__) && (__clang_major__ >= 23)
  i32x8 g1b = {};
  __builtin_amdgcn_tensor_load_to_lds(g0, g1, g2, g3, g1b, 0);
#else
  __builtin_amdgcn_tensor_load_to_lds(g0, g1, g2, g3, 0);
#endif
}
#else
#define HAVE_TDM 0
#endif

// ---- WMMA GEMM with branch-free double-buffered async staging --------------
// C[M,N] = act(A[M,K] @ W[N,K]^T + bias); act: 0=none, 1=tanh. K % 32 == 0.
// OOB rows are staged via index clamping: they only feed OOB C rows/cols,
// which the guarded writeback discards.

__global__ void __launch_bounds__(256)
k_wmma_gemm(const _Float16* __restrict__ A, const _Float16* __restrict__ W,
            const float* __restrict__ bias, float* __restrict__ C32,
            _Float16* __restrict__ C16, int M, int N, int K, int act)
{
  __shared__ _Float16 As[2][WMMA_BM * LDT];
  __shared__ _Float16 Ws[2][WMMA_BN * LDT];

  const int tid  = threadIdx.x;
  const int wave = tid >> 5;
  const int lane = tid & 31;
  const int lh   = lane >> 4;
  const int lm   = lane & 15;
  const int wm   = wave >> 1;   // 0..3 -> 16 M-rows each
  const int wn   = wave & 1;    // 0..1 -> 64 N-cols each

  const int blockM = blockIdx.y * WMMA_BM;
  const int blockN = blockIdx.x * WMMA_BN;

  v8f acc[4] = {};

  // A staging: 64x32 halves -> one b128 per thread
  const int ar_ = tid >> 2;             // 0..63
  const int ac_ = (tid & 3) * 8;        // 0,8,16,24
  const size_t abase = (size_t)imin_(blockM + ar_, M - 1) * K;
  const unsigned aoff = (unsigned)(uintptr_t)&As[0][ar_ * LDT + ac_];
  // W staging: 128x32 halves -> two b128 per thread
  const int wr_ = tid >> 1;             // 0..127
  const int wc_ = (tid & 1) * 16;       // 0,16
  const size_t wbase = (size_t)imin_(blockN + wr_, N - 1) * K;
  const unsigned woff = (unsigned)(uintptr_t)&Ws[0][wr_ * LDT + wc_];

  const unsigned ABUF = (unsigned)(sizeof(_Float16) * WMMA_BM * LDT);
  const unsigned WBUF = (unsigned)(sizeof(_Float16) * WMMA_BN * LDT);

  // prologue: stage tile 0 into buffer 0
  async_ld_b128(aoff, A + abase + ac_);
  async_ld_b128(woff, W + wbase + wc_);
  async_ld_b128(woff + 16, W + wbase + wc_ + 8);

  const int nk = K / WMMA_BK;
  for (int it = 0; it < nk; ++it) {
    wait_async0();
    __syncthreads();
    // overlap: DMA tile it+1 while computing tile it
    if (it + 1 < nk) {
      const int kn = (it + 1) * WMMA_BK;
      const unsigned nb = (unsigned)((it + 1) & 1);
      async_ld_b128(aoff + nb * ABUF, A + abase + kn + ac_);
      async_ld_b128(woff + nb * WBUF, W + wbase + kn + wc_);
      async_ld_b128(woff + nb * WBUF + 16, W + wbase + kn + wc_ + 8);
    }
    const _Float16* as = As[it & 1];
    const _Float16* ws = Ws[it & 1];
    v16h af = load_a_frag(&as[(wm * 16 + lm) * LDT], lh);
#pragma unroll
    for (int sub = 0; sub < 4; ++sub) {
      v16h bf = load_b_frag(&ws[(wn * 64 + sub * 16 + lm) * LDT], lh);
      acc[sub] = __builtin_amdgcn_wmma_f32_16x16x32_f16(false, af, false, bf,
                                                        (short)0, acc[sub], false, false);
    }
  }

#pragma unroll
  for (int sub = 0; sub < 4; ++sub) {
    const int cn = blockN + wn * 64 + sub * 16 + lm;
    if (cn < N) {
      const float bv = bias ? bias[cn] : 0.f;
#pragma unroll
      for (int i = 0; i < 8; ++i) {
        const int cm = blockM + wm * 16 + lh * 8 + i;
        if (cm < M) {
          float v = acc[sub][i] + bv;
          if (act == 1) v = tanhf(v);
          const size_t o = (size_t)cm * N + cn;
          if (C32) C32[o] = v;
          if (C16) C16[o] = (_Float16)v;
        }
      }
    }
  }
}

// ---- WMMA GEMM with double-buffered TDM staging ----------------------------

__global__ void __launch_bounds__(256)
k_wmma_gemm_tdm(const _Float16* __restrict__ A, const _Float16* __restrict__ W,
                const float* __restrict__ bias, float* __restrict__ C32,
                _Float16* __restrict__ C16, int M, int N, int K, int act)
{
  __shared__ _Float16 As[2][WMMA_BM * LDT];
  __shared__ _Float16 Ws[2][WMMA_BN * LDT];

  const int tid  = threadIdx.x;
  const int wave = tid >> 5;
  const int lane = tid & 31;
  const int lh   = lane >> 4;
  const int lm   = lane & 15;
  const int wm   = wave >> 1;
  const int wn   = wave & 1;

  const int blockM = blockIdx.y * WMMA_BM;
  const int blockN = blockIdx.x * WMMA_BN;

  v8f acc[4] = {};

  const int nk = K / WMMA_BK;

#if HAVE_TDM
  const unsigned asAddr = (unsigned)(uintptr_t)&As[0][0];
  const unsigned wsAddr = (unsigned)(uintptr_t)&Ws[0][0];
  const unsigned ABUF = (unsigned)(sizeof(_Float16) * WMMA_BM * LDT);
  const unsigned WBUF = (unsigned)(sizeof(_Float16) * WMMA_BN * LDT);
  if (tid < 32) {   // wave 0 drives the Tensor Data Mover (EXEC-independent DMA)
    tdm_load_tile(asAddr, A + (size_t)blockM * K, K, M - blockM, K, WMMA_BM);
    tdm_load_tile(wsAddr, W + (size_t)blockN * K, K, N - blockN, K, WMMA_BN);
  }
#else
  // fallback: branch-free per-lane async staging (clamped rows)
  const int ar_ = tid >> 2;
  const int ac_ = (tid & 3) * 8;
  const size_t abase = (size_t)imin_(blockM + ar_, M - 1) * K;
  const unsigned aoff = (unsigned)(uintptr_t)&As[0][ar_ * LDT + ac_];
  const int wr_ = tid >> 1;
  const int wc_ = (tid & 1) * 16;
  const size_t wbase = (size_t)imin_(blockN + wr_, N - 1) * K;
  const unsigned woff = (unsigned)(uintptr_t)&Ws[0][wr_ * LDT + wc_];
  const unsigned ABUF = (unsigned)(sizeof(_Float16) * WMMA_BM * LDT);
  const unsigned WBUF = (unsigned)(sizeof(_Float16) * WMMA_BN * LDT);
  async_ld_b128(aoff, A + abase + ac_);
  async_ld_b128(woff, W + wbase + wc_);
  async_ld_b128(woff + 16, W + wbase + wc_ + 8);
#endif

  for (int it = 0; it < nk; ++it) {
#if HAVE_TDM
    if (tid < 32) __builtin_amdgcn_s_wait_tensorcnt((short)0);
#else
    wait_async0();
#endif
    __syncthreads();
    if (it + 1 < nk) {
      const int kn = (it + 1) * WMMA_BK;
      const unsigned nb = (unsigned)((it + 1) & 1);
#if HAVE_TDM
      if (tid < 32) {
        tdm_load_tile(asAddr + nb * ABUF, A + (size_t)blockM * K + kn,
                      K - kn, M - blockM, K, WMMA_BM);
        tdm_load_tile(wsAddr + nb * WBUF, W + (size_t)blockN * K + kn,
                      K - kn, N - blockN, K, WMMA_BN);
      }
#else
      async_ld_b128(aoff + nb * ABUF, A + abase + kn + ac_);
      async_ld_b128(woff + nb * WBUF, W + wbase + kn + wc_);
      async_ld_b128(woff + nb * WBUF + 16, W + wbase + kn + wc_ + 8);
#endif
    }
    const _Float16* as = As[it & 1];
    const _Float16* ws = Ws[it & 1];
    v16h af = load_a_frag(&as[(wm * 16 + lm) * LDT], lh);
#pragma unroll
    for (int sub = 0; sub < 4; ++sub) {
      v16h bf = load_b_frag(&ws[(wn * 64 + sub * 16 + lm) * LDT], lh);
      acc[sub] = __builtin_amdgcn_wmma_f32_16x16x32_f16(false, af, false, bf,
                                                        (short)0, acc[sub], false, false);
    }
  }

#pragma unroll
  for (int sub = 0; sub < 4; ++sub) {
    const int cn = blockN + wn * 64 + sub * 16 + lm;
    if (cn < N) {
      const float bv = bias ? bias[cn] : 0.f;
#pragma unroll
      for (int i = 0; i < 8; ++i) {
        const int cm = blockM + wm * 16 + lh * 8 + i;
        if (cm < M) {
          float v = acc[sub][i] + bv;
          if (act == 1) v = tanhf(v);
          const size_t o = (size_t)cm * N + cn;
          if (C32) C32[o] = v;
          if (C16) C16[o] = (_Float16)v;
        }
      }
    }
  }
}

// ---- LSTM pointwise cell update (PyTorch gate order i,f,g,o) ---------------

__global__ void __launch_bounds__(256)
k_lstm_pointwise(const float* __restrict__ gg, const float* __restrict__ xproj,
                 float* __restrict__ cst, _Float16* __restrict__ h16,
                 _Float16* __restrict__ hs, int t, int Tt, int Bn, int Hn)
{
  const int i = blockIdx.x * blockDim.x + threadIdx.x;
  if (i >= Bn * Hn) return;
  const int b = i / Hn, j = i % Hn;
  const size_t gb = (size_t)b * 4 * Hn + j;
  const size_t xb = ((size_t)b * Tt + t) * 4 * Hn + j;
  const float ig = gg[gb]          + xproj[xb];
  const float fg = gg[gb + Hn]     + xproj[xb + Hn];
  const float cg = gg[gb + 2 * Hn] + xproj[xb + 2 * Hn];
  const float og = gg[gb + 3 * Hn] + xproj[xb + 3 * Hn];
  const float c  = sigmoidf_(fg) * cst[i] + sigmoidf_(ig) * tanhf(cg);
  const float h  = sigmoidf_(og) * tanhf(c);
  cst[i] = c;
  h16[i] = (_Float16)h;
  hs[((size_t)b * Tt + t) * Hn + j] = (_Float16)h;
}

// ---- time reduction --------------------------------------------------------

__global__ void __launch_bounds__(256)
k_time_reduce(const _Float16* __restrict__ in, _Float16* __restrict__ out,
              int Bn, int Tin, int Tout, int D)
{
  const int i = blockIdx.x * blockDim.x + threadIdx.x;
  if (i >= Bn * Tout * D) return;
  const int d = i % D;
  const int r = i / D;
  const int t = r % Tout, b = r / Tout;
  const size_t s = ((size_t)b * Tin + 2 * t) * D + d;
  out[i] = (_Float16)(0.5f * ((float)in[s] + (float)in[s + D]));
}

// ---- LayerNorm -------------------------------------------------------------

__global__ void __launch_bounds__(256)
k_layernorm(const float* __restrict__ x, const float* __restrict__ gam,
            const float* __restrict__ bet, _Float16* __restrict__ out, int D)
{
  __shared__ float red[256];
  const int row = blockIdx.x;
  const float* xr = x + (size_t)row * D;

  float s = 0.f;
  for (int i = threadIdx.x; i < D; i += 256) s += xr[i];
  red[threadIdx.x] = s; __syncthreads();
  for (int st = 128; st > 0; st >>= 1) {
    if (threadIdx.x < st) red[threadIdx.x] += red[threadIdx.x + st];
    __syncthreads();
  }
  const float mu = red[0] / (float)D;
  __syncthreads();

  float v = 0.f;
  for (int i = threadIdx.x; i < D; i += 256) { float d = xr[i] - mu; v += d * d; }
  red[threadIdx.x] = v; __syncthreads();
  for (int st = 128; st > 0; st >>= 1) {
    if (threadIdx.x < st) red[threadIdx.x] += red[threadIdx.x + st];
    __syncthreads();
  }
  const float rstd = rsqrtf(red[0] / (float)D + 1e-5f);

  for (int i = threadIdx.x; i < D; i += 256)
    out[(size_t)row * D + i] = (_Float16)((xr[i] - mu) * rstd * gam[i] + bet[i]);
}

// ---- embedding gather (BOS prepended; PAD row of embed is zero) ------------

__global__ void __launch_bounds__(256)
k_embed_gather(const int* __restrict__ ys, const float* __restrict__ embed,
               _Float16* __restrict__ out, int Bn, int Un, int U1n, int En)
{
  const int i = blockIdx.x * blockDim.x + threadIdx.x;
  if (i >= Bn * U1n * En) return;
  const int e = i % En;
  const int r = i / En;
  const int u = r % U1n, b = r / U1n;
  const int tok = (u == 0) ? 1 /*BOS*/ : ys[b * Un + (u - 1)];
  out[i] = (_Float16)embed[(size_t)tok * En + e];
}

// ---- joint broadcast-add ---------------------------------------------------

__global__ void __launch_bounds__(256)
k_joint_add(const _Float16* __restrict__ henc, const _Float16* __restrict__ hpre,
            _Float16* __restrict__ z, int Bn, int Tn, int U1n, int Pn)
{
  const int i = blockIdx.x * blockDim.x + threadIdx.x;
  if (i >= Bn * Tn * U1n * Pn) return;
  const int p = i % Pn;
  int r = i / Pn;
  const int u = r % U1n; r /= U1n;
  const int t = r % Tn;  const int b = r / Tn;
  z[i] = (_Float16)((float)henc[((size_t)b * Tn + t) * Pn + p] +
                    (float)hpre[((size_t)b * U1n + u) * Pn + p]);
}

// ---------------------------------------------------------------------------

extern "C" void kernel_launch(void* const* d_in, const int* in_sizes, int n_in,
                              void* d_out, int out_size, void* d_ws, size_t ws_size,
                              hipStream_t stream) {
  (void)in_sizes; (void)n_in; (void)out_size; (void)ws_size;
  constexpr int B = 4, T = 256, U = 64, F = 80, H = 2048, P = 640, V = 1024, E = 256;
  constexpr int U1 = U + 1, T2 = T / 2, G = 4 * H;
  constexpr int FP = 96;   // layer-0 K padded to a multiple of 32

  const float* xs = (const float*)d_in[0];
  const int*   ys = (const int*)  d_in[1];

  struct Layer { const float *Wih, *Whh, *bih, *bhh, *Wp, *bp, *gam, *bet; int kin, kpad; };
  Layer enc[8], dec[2];
  int pi = 2;
  for (int l = 0; l < 8; ++l, pi += 8) {
    enc[l] = { (const float*)d_in[pi], (const float*)d_in[pi + 1],
               (const float*)d_in[pi + 2], (const float*)d_in[pi + 3],
               (const float*)d_in[pi + 4], (const float*)d_in[pi + 5],
               (const float*)d_in[pi + 6], (const float*)d_in[pi + 7],
               (l == 0) ? F : P, (l == 0) ? FP : P };
  }
  for (int l = 0; l < 2; ++l, pi += 8) {
    dec[l] = { (const float*)d_in[pi], (const float*)d_in[pi + 1],
               (const float*)d_in[pi + 2], (const float*)d_in[pi + 3],
               (const float*)d_in[pi + 4], (const float*)d_in[pi + 5],
               (const float*)d_in[pi + 6], (const float*)d_in[pi + 7],
               (l == 0) ? E : P, (l == 0) ? E : P };
  }
  const float* embed = (const float*)d_in[pi++];
  const float* jw1   = (const float*)d_in[pi++];
  const float* jb1   = (const float*)d_in[pi++];
  const float* jw2   = (const float*)d_in[pi++];
  const float* jb2   = (const float*)d_in[pi++];

  // scratch arena
  size_t off = 0;
  auto arena = [&](size_t bytes) -> void* {
    size_t a = (off + 255) & ~(size_t)255;
    off = a + bytes;
    return (void*)((char*)d_ws + a);
  };
  _Float16* wih16 = (_Float16*)arena((size_t)G * P * 2);
  _Float16* whh16 = (_Float16*)arena((size_t)G * H * 2);
  _Float16* wp16  = (_Float16*)arena((size_t)P * H * 2);
  float*    bsum  = (float*)   arena((size_t)G * 4);
  _Float16* xin16 = (_Float16*)arena((size_t)B * T * P * 2);
  float*    xproj = (float*)   arena((size_t)B * T * G * 4);
  float*    ggem  = (float*)   arena((size_t)B * G * 4);
  _Float16* hset  = (_Float16*)arena((size_t)B * H * 2);
  float*    cset  = (float*)   arena((size_t)B * H * 4);
  _Float16* hs16  = (_Float16*)arena((size_t)B * T * H * 2);
  _Float16* hred  = (_Float16*)arena((size_t)B * T2 * H * 2);
  float*    prj32 = (float*)   arena((size_t)B * T * P * 4);
  _Float16* henc  = (_Float16*)arena((size_t)B * T2 * P * 2);
  _Float16* hpre  = (_Float16*)arena((size_t)B * U1 * P * 2);
  _Float16* emb16 = (_Float16*)arena((size_t)B * U1 * E * 2);
  _Float16* z16   = (_Float16*)arena((size_t)B * T2 * U1 * P * 2);
  _Float16* h1_16 = (_Float16*)arena((size_t)B * T2 * U1 * P * 2);
  _Float16* jw116 = (_Float16*)arena((size_t)P * P * 2);
  _Float16* jw216 = (_Float16*)arena((size_t)V * P * 2);

  auto cvt = [&](const float* s, _Float16* d, long n) {
    k_cvt_f32_f16<<<dim3((unsigned)((n + 255) / 256)), dim3(256), 0, stream>>>(s, d, (int)n);
  };
  auto gemm = [&](const _Float16* A, const _Float16* W, const float* bias,
                  float* C32, _Float16* C16, int M, int N, int K, int act) {
    dim3 g((N + WMMA_BN - 1) / WMMA_BN, (M + WMMA_BM - 1) / WMMA_BM);
    k_wmma_gemm<<<g, dim3(256), 0, stream>>>(A, W, bias, C32, C16, M, N, K, act);
  };
  auto gemm_tdm = [&](const _Float16* A, const _Float16* W, const float* bias,
                      float* C32, _Float16* C16, int M, int N, int K, int act) {
    dim3 g((N + WMMA_BN - 1) / WMMA_BN, (M + WMMA_BM - 1) / WMMA_BM);
    k_wmma_gemm_tdm<<<g, dim3(256), 0, stream>>>(A, W, bias, C32, C16, M, N, K, act);
  };

  // projected-LSTM stack (encoder/decoder share structure)
  auto run_stack = [&](const _Float16* in0, int Tin, const Layer* Ls, int nl,
                       int red_after, _Float16* final16) {
    const _Float16* cur = in0;
    int curT = Tin;
    for (int l = 0; l < nl; ++l) {
      const Layer& Lp = Ls[l];
      if (Lp.kpad != Lp.kin) {
        k_cvt_pad_f16<<<dim3((G * Lp.kpad + 255) / 256), dim3(256), 0, stream>>>(
            Lp.Wih, wih16, G, Lp.kin, Lp.kpad);
      } else {
        cvt(Lp.Wih, wih16, (long)G * Lp.kin);
      }
      cvt(Lp.Whh, whh16, (long)G * H);
      cvt(Lp.Wp,  wp16,  (long)P * H);
      k_add2_f32<<<dim3((G + 255) / 256), dim3(256), 0, stream>>>(Lp.bih, Lp.bhh, bsum, G);
      // x_proj for all timesteps (TDM-staged GEMM)
      gemm_tdm(cur, wih16, bsum, xproj, nullptr, B * curT, G, Lp.kpad, 0);
      k_zero_f16<<<dim3((B * H + 255) / 256), dim3(256), 0, stream>>>(hset, B * H);
      k_zero_f32<<<dim3((B * H + 255) / 256), dim3(256), 0, stream>>>(cset, B * H);
      // recurrence: per-step (B,H)@Whh^T on the matrix pipe (async-staged GEMM)
      for (int t = 0; t < curT; ++t) {
        gemm(hset, whh16, nullptr, ggem, nullptr, B, G, H, 0);
        k_lstm_pointwise<<<dim3((B * H + 255) / 256), dim3(256), 0, stream>>>(
            ggem, xproj, cset, hset, hs16, t, curT, B, H);
      }
      const _Float16* seq = hs16;
      if (l == red_after) {
        const int Tout = curT / 2;
        k_time_reduce<<<dim3((B * Tout * H + 255) / 256), dim3(256), 0, stream>>>(
            hs16, hred, B, curT, Tout, H);
        seq = hred; curT = Tout;
      }
      gemm_tdm(seq, wp16, Lp.bp, prj32, nullptr, B * curT, P, H, 0);
      _Float16* lnout = (l == nl - 1) ? final16 : xin16;
      k_layernorm<<<dim3(B * curT), dim3(256), 0, stream>>>(prj32, Lp.gam, Lp.bet, lnout, P);
      cur = lnout;
    }
  };

  // encoder (input padded 80 -> 96 so every GEMM K is a multiple of 32)
  k_cvt_pad_f16<<<dim3((B * T * FP + 255) / 256), dim3(256), 0, stream>>>(
      xs, xin16, B * T, F, FP);
  run_stack(xin16, T, enc, 8, /*red_after=*/1, henc);

  // decoder (prediction network)
  k_embed_gather<<<dim3((B * U1 * E + 255) / 256), dim3(256), 0, stream>>>(
      ys, embed, emb16, B, U, U1, E);
  run_stack(emb16, U1, dec, 2, /*red_after=*/-1, hpre);

  // joint network
  k_joint_add<<<dim3((B * T2 * U1 * P + 255) / 256), dim3(256), 0, stream>>>(
      henc, hpre, z16, B, T2, U1, P);
  cvt(jw1, jw116, (long)P * P);
  cvt(jw2, jw216, (long)V * P);
  gemm_tdm(z16,   jw116, jb1, nullptr, h1_16, B * T2 * U1, P, P, /*tanh*/1);
  gemm_tdm(h1_16, jw216, jb2, (float*)d_out, nullptr, B * T2 * U1, V, P, 0);
}
